// CrossViewBEVLifter_62878321214192
// MI455X (gfx1250) — matrix operand order repaired and checked by reading, not compiled, not run on hardware
//
#include <hip/hip_runtime.h>
#include <hip/hip_bf16.h>

typedef __attribute__((ext_vector_type(16))) _Float16 v16h;
typedef __attribute__((ext_vector_type(8)))  _Float16 v8h;
typedef __attribute__((ext_vector_type(8)))  float    v8f;

#define BEV_H 320
#define BEV_W 320
#define NCELL (BEV_H * BEV_W)
#define FEAT_H 128
#define FEAT_W 352
#define CCH 96
#define HID 64
#define NB 2
#define NH 3
#define BN_EPS 1e-3f

// ---------------------------------------------------------------------------
// K0: pack conv weights (BN-scale folded, f16, WMMA A-frag layout) + sh_w1
// (WMMA B-frag layout) + fused BN biases.
// A-frag (16x32 f16): lane L -> M = L%16, s = L/16; elem e -> K = s*16 + e
// B-frag (32x16 f16): lane L -> N = L%16, s = L/16; elem e -> K = (e/8)*16 + s*8 + e%8
// ---------------------------------------------------------------------------
__global__ void prep_kernel(const float* __restrict__ w1, const float* __restrict__ g1,
                            const float* __restrict__ b1, const float* __restrict__ m1,
                            const float* __restrict__ v1,
                            const float* __restrict__ w2, const float* __restrict__ g2,
                            const float* __restrict__ b2, const float* __restrict__ m2,
                            const float* __restrict__ v2,
                            const float* __restrict__ sw1,
                            _Float16* __restrict__ w1p, float* __restrict__ bias1,
                            _Float16* __restrict__ w2p, float* __restrict__ bias2,
                            _Float16* __restrict__ sw1p)
{
    int idx = blockIdx.x * blockDim.x + threadIdx.x;
    // conv weight packing: 6 octiles x 27 ksteps x 32 lanes x 16 elems = 82944
    if (idx < 6 * 27 * 512) {
        int octile = idx / (27 * 512);
        int rem    = idx % (27 * 512);
        int ks     = rem / 512;
        int li     = rem % 512;
        int lane   = li / 16;
        int e      = li % 16;
        int oc = octile * 16 + (lane & 15);
        int s  = lane >> 4;
        int kk = s * 16 + e;
        int K  = ks * 32 + kk;           // K = tap*96 + ci
        int tap = K / CCH, ci = K % CCH;
        int ky = tap / 3, kx = tap % 3;
        int widx = ((oc * CCH + ci) * 3 + ky) * 3 + kx;
        float sc1 = g1[oc] * rsqrtf(v1[oc] + BN_EPS);
        float sc2 = g2[oc] * rsqrtf(v2[oc] + BN_EPS);
        w1p[idx] = (_Float16)(w1[widx] * sc1);
        w2p[idx] = (_Float16)(w2[widx] * sc2);
    }
    // sh_w1 packing: 3 ksteps x 4 ntiles x 32 lanes x 16 elems = 6144
    if (idx < 3 * 4 * 512) {
        int kt   = idx / 512;           // 0..11
        int ks   = kt / 4;
        int nt   = kt % 4;
        int li   = idx % 512;
        int lane = li / 16;
        int e    = li % 16;
        int c = ks * 32 + (e / 8) * 16 + (lane >> 4) * 8 + (e % 8);   // K index
        int j = nt * 16 + (lane & 15);            // hidden index
        sw1p[idx] = (_Float16)sw1[c * HID + j];
    }
    if (idx < CCH) {
        float sc1 = g1[idx] * rsqrtf(v1[idx] + BN_EPS);
        float sc2 = g2[idx] * rsqrtf(v2[idx] + BN_EPS);
        bias1[idx] = b1[idx] - m1[idx] * sc1;
        bias2[idx] = b2[idx] - m2[idx] * sc2;
    }
}

// ---------------------------------------------------------------------------
// K1: projection + bilinear sample + depth MLP + height embedding.
// One wave per (b, cell, h); 3 channels per lane (c = lane, lane+32, lane+64).
// ---------------------------------------------------------------------------
__global__ void lift_sample_kernel(const float* __restrict__ feat,
                                   const float* __restrict__ l2i,
                                   const float* __restrict__ he,
                                   const float* __restrict__ dw1,
                                   const float* __restrict__ db1,
                                   const float* __restrict__ dw2,
                                   const float* __restrict__ db2,
                                   _Float16* __restrict__ tokens,
                                   _Float16* __restrict__ sampled,
                                   float* __restrict__ validf)
{
    int unit = blockIdx.x * (blockDim.x >> 5) + (threadIdx.x >> 5);
    int lane = threadIdx.x & 31;
    const int NUNITS = NB * NCELL * NH;
    if (unit >= NUNITS) return;

    int h    = unit % NH;
    int cell = unit / NH;
    int n = cell % NCELL;
    int b = cell / NCELL;
    int iy = n / BEV_W, ix = n % BEV_W;

    float px = ((float)ix + 0.5f) * (102.4f / (float)BEV_W) - 51.2f;
    float py = ((float)iy + 0.5f) * (102.4f / (float)BEV_H) - 51.2f;
    float pz = (float)h;                       // HEIGHTS = 0,1,2

    const float* M = l2i + b * 16;
    float p0 = M[0] * px + M[1] * py + M[2]  * pz + M[3];
    float p1 = M[4] * px + M[5] * py + M[6]  * pz + M[7];
    float p2 = M[8] * px + M[9] * py + M[10] * pz + M[11];

    float depth = p2;
    float dcl = fmaxf(depth, 1e-5f);
    float u = p0 / dcl, v = p1 / dcl;
    bool valid = (depth > 1e-3f) && (u >= 0.f) && (u <= (float)(FEAT_W - 1))
                                 && (v >= 0.f) && (v <= (float)(FEAT_H - 1));
    float vmask = valid ? 1.f : 0.f;

    float xs = fminf(fmaxf(u, -8.f), (float)FEAT_W + 8.f);
    float ys = fminf(fmaxf(v, -8.f), (float)FEAT_H + 8.f);
    float x0f = floorf(xs), y0f = floorf(ys);
    int   x0 = (int)x0f,  y0 = (int)y0f;
    float wx = xs - x0f,  wy = ys - y0f;
    float w00 = (1.f - wx) * (1.f - wy), w10 = wx * (1.f - wy);
    float w01 = (1.f - wx) * wy,         w11 = wx * wy;
    bool bx0 = (x0 >= 0) && (x0 <= FEAT_W - 1);
    bool bx1 = (x0 + 1 >= 0) && (x0 + 1 <= FEAT_W - 1);
    bool by0 = (y0 >= 0) && (y0 <= FEAT_H - 1);
    bool by1 = (y0 + 1 >= 0) && (y0 + 1 <= FEAT_H - 1);
    int xc0 = min(max(x0, 0), FEAT_W - 1), xc1 = min(max(x0 + 1, 0), FEAT_W - 1);
    int yc0 = min(max(y0, 0), FEAT_H - 1), yc1 = min(max(y0 + 1, 0), FEAT_H - 1);

    const float* fb = feat + (size_t)b * CCH * FEAT_H * FEAT_W;
    float d = log1pf(fmaxf(depth, 1e-3f));

    int c0 = lane, c1 = lane + 32, c2 = lane + 64;

    float sv[3];
    int cs[3] = {c0, c1, c2};
    #pragma unroll
    for (int k = 0; k < 3; ++k) {
        const float* fc = fb + (size_t)cs[k] * FEAT_H * FEAT_W;
        float acc = 0.f;
        if (bx0 && by0) acc += fc[yc0 * FEAT_W + xc0] * w00;
        if (bx1 && by0) acc += fc[yc0 * FEAT_W + xc1] * w10;
        if (bx0 && by1) acc += fc[yc1 * FEAT_W + xc0] * w01;
        if (bx1 && by1) acc += fc[yc1 * FEAT_W + xc1] * w11;
        sv[k] = acc * vmask;
    }

    // depth MLP: hidden_j depends only on scalar d -> regenerate per lane
    float a0 = db2[c0], a1 = db2[c1], a2 = db2[c2];
    for (int j = 0; j < HID; ++j) {
        float hj = fmaxf(d * dw1[j] + db1[j], 0.f);
        a0 += hj * dw2[j * CCH + c0];
        a1 += hj * dw2[j * CCH + c1];
        a2 += hj * dw2[j * CCH + c2];
    }

    size_t row = (size_t)unit * CCH;
    tokens[row + c0] = (_Float16)(sv[0] + a0 + he[h * CCH + c0]);
    tokens[row + c1] = (_Float16)(sv[1] + a1 + he[h * CCH + c1]);
    tokens[row + c2] = (_Float16)(sv[2] + a2 + he[h * CCH + c2]);
    sampled[row + c0] = (_Float16)sv[0];
    sampled[row + c1] = (_Float16)sv[1];
    sampled[row + c2] = (_Float16)sv[2];
    if (lane == 0) validf[unit] = vmask;
}

// ---------------------------------------------------------------------------
// K2: score MLP via WMMA.  hidden = relu(token @ sw1 + sb1); logit = hidden@sw2+sb2
// One wave per 16-token tile. 4 N-tiles x 3 K-steps of v_wmma_f32_16x16x32_f16.
// ---------------------------------------------------------------------------
__global__ void score_kernel(const _Float16* __restrict__ tokens,
                             const _Float16* __restrict__ sw1p,
                             const float* __restrict__ sb1,
                             const float* __restrict__ sw2,
                             const float* __restrict__ sb2,
                             float* __restrict__ logits)
{
    int wave = blockIdx.x * (blockDim.x >> 5) + (threadIdx.x >> 5);
    int lane = threadIdx.x & 31;
    const int NT = (NB * NCELL * NH) / 16;   // 38400 tiles, exact
    if (wave >= NT) return;

    int rowbase = wave * 16;
    int mrow = rowbase + (lane & 15);
    int s = lane >> 4;

    v16h a[3];
    #pragma unroll
    for (int ks = 0; ks < 3; ++ks) {
        const _Float16* tp = tokens + (size_t)mrow * CCH + ks * 32 + s * 8;
        v8h lo = *(const v8h*)tp;
        v8h hi = *(const v8h*)(tp + 16);
        a[ks] = __builtin_shufflevector(lo, hi, 0, 1, 2, 3, 4, 5, 6, 7,
                                                8, 9, 10, 11, 12, 13, 14, 15);
    }

    float part[8] = {0.f, 0.f, 0.f, 0.f, 0.f, 0.f, 0.f, 0.f};
    #pragma unroll
    for (int nt = 0; nt < 4; ++nt) {
        v8f acc = {};
        #pragma unroll
        for (int ks = 0; ks < 3; ++ks) {
            v16h bfrag = *(const v16h*)(sw1p + ((size_t)(ks * 4 + nt) * 32 + lane) * 16);
            acc = __builtin_amdgcn_wmma_f32_16x16x32_f16(
                false, a[ks], false, bfrag, (short)0, acc, false, false);
        }
        int j = nt * 16 + (lane & 15);
        float bj = sb1[j], s2 = sw2[j];
        #pragma unroll
        for (int r = 0; r < 8; ++r)
            part[r] += fmaxf(acc[r] + bj, 0.f) * s2;
    }

    // reduce over the 16 lanes that share a token (xor net stays within halves)
    #pragma unroll
    for (int r = 0; r < 8; ++r) {
        float p = part[r];
        p += __shfl_xor(p, 1, 32);
        p += __shfl_xor(p, 2, 32);
        p += __shfl_xor(p, 4, 32);
        p += __shfl_xor(p, 8, 32);
        part[r] = p + sb2[0];
    }
    if (lane == 0) {
        #pragma unroll
        for (int r = 0; r < 8; ++r) logits[rowbase + r] = part[r];
    }
    if (lane == 16) {
        #pragma unroll
        for (int r = 0; r < 8; ++r) logits[rowbase + 8 + r] = part[r];
    }
}

// ---------------------------------------------------------------------------
// K3: masked softmax over 3 heights + weighted fuse -> BEV NHWC f16.
// ---------------------------------------------------------------------------
__global__ void fuse_kernel(const float* __restrict__ logits,
                            const float* __restrict__ validf,
                            const _Float16* __restrict__ sampled,
                            _Float16* __restrict__ bev)
{
    int idx = blockIdx.x * blockDim.x + threadIdx.x;
    const int TOT = NB * NCELL * CCH;
    if (idx >= TOT) return;
    int c = idx % CCH;
    int cell = idx / CCH;

    float v0 = validf[cell * 3 + 0];
    float v1 = validf[cell * 3 + 1];
    float v2 = validf[cell * 3 + 2];
    float l0 = (v0 > 0.f) ? logits[cell * 3 + 0] : -1e30f;
    float l1 = (v1 > 0.f) ? logits[cell * 3 + 1] : -1e30f;
    float l2 = (v2 > 0.f) ? logits[cell * 3 + 2] : -1e30f;
    bool hv = (v0 + v1 + v2) > 0.5f;
    float m = fmaxf(fmaxf(l0, l1), l2);
    float e0 = (v0 > 0.f) ? expf(l0 - m) : 0.f;
    float e1 = (v1 > 0.f) ? expf(l1 - m) : 0.f;
    float e2 = (v2 > 0.f) ? expf(l2 - m) : 0.f;
    float Z = e0 + e1 + e2;
    float inv = hv ? (1.f / Z) : 0.f;

    float s0 = (float)sampled[(size_t)(cell * 3 + 0) * CCH + c];
    float s1 = (float)sampled[(size_t)(cell * 3 + 1) * CCH + c];
    float s2 = (float)sampled[(size_t)(cell * 3 + 2) * CCH + c];
    float fused = (e0 * s0 + e1 * s1 + e2 * s2) * inv;
    bev[idx] = (_Float16)fused;   // idx == (b*NCELL+n)*96+c == NHWC
}

// ---------------------------------------------------------------------------
// K4: conv1 3x3 (implicit GEMM, WMMA) + BN + ReLU.  NHWC f16 -> NHWC f16.
// One wave per 16-pixel tile, ALL 6 out-channel tiles kept in registers:
// each im2col B fragment is loaded once and feeds 6 back-to-back WMMAs.
// ---------------------------------------------------------------------------
__global__ void conv1_kernel(const _Float16* __restrict__ in,
                             const _Float16* __restrict__ wp,
                             const float* __restrict__ bias,
                             _Float16* __restrict__ out)
{
    int pt   = blockIdx.x * (blockDim.x >> 5) + (threadIdx.x >> 5);
    int lane = threadIdx.x & 31;
    const int NU = NB * BEV_H * (BEV_W / 16);   // 12800 pixel tiles
    if (pt >= NU) return;

    int b = pt / (BEV_H * (BEV_W / 16));
    int rem = pt % (BEV_H * (BEV_W / 16));
    int y = rem / (BEV_W / 16);
    int x0 = (rem % (BEV_W / 16)) * 16;
    int lx = lane & 15, s = lane >> 4;
    int x = x0 + lx;

    v8f acc[6];
    #pragma unroll
    for (int o = 0; o < 6; ++o) acc[o] = v8f{};

    #pragma unroll
    for (int ks = 0; ks < 27; ++ks) {
        int tap = ks / 3, cib = (ks % 3) * 32;
        int ky = tap / 3, kx = tap % 3;
        int yy = y + ky - 1, xx = x + kx - 1;
        v16h bfrag = {};
        if (yy >= 0 && yy < BEV_H && xx >= 0 && xx < BEV_W)
            bfrag = *(const v16h*)(in + (((size_t)(b * BEV_H + yy) * BEV_W + xx) * CCH
                                         + cib + s * 16));
        #pragma unroll
        for (int o = 0; o < 6; ++o) {
            v16h afrag = *(const v16h*)(wp + ((size_t)(o * 27 + ks) * 32 + lane) * 16);
            acc[o] = __builtin_amdgcn_wmma_f32_16x16x32_f16(
                false, afrag, false, bfrag, (short)0, acc[o], false, false);
        }
    }

    size_t pixbase = ((size_t)(b * BEV_H + y) * BEV_W + x) * CCH;
    #pragma unroll
    for (int o = 0; o < 6; ++o) {
        int ocb = o * 16 + s * 8;
        v8h ov;
        #pragma unroll
        for (int r = 0; r < 8; ++r)
            ov[r] = (_Float16)fmaxf(acc[o][r] + bias[ocb + r], 0.f);
        *(v8h*)(out + pixbase + ocb) = ov;
    }
}

// ---------------------------------------------------------------------------
// K5: conv2 3x3 (implicit GEMM, WMMA) + BN + ReLU.  NHWC f16 -> NCHW f32 out.
// Same all-outC-per-wave structure; epilogue writes the final f32 NCHW output.
// ---------------------------------------------------------------------------
__global__ void conv2_kernel(const _Float16* __restrict__ in,
                             const _Float16* __restrict__ wp,
                             const float* __restrict__ bias,
                             float* __restrict__ out)
{
    int pt   = blockIdx.x * (blockDim.x >> 5) + (threadIdx.x >> 5);
    int lane = threadIdx.x & 31;
    const int NU = NB * BEV_H * (BEV_W / 16);   // 12800 pixel tiles
    if (pt >= NU) return;

    int b = pt / (BEV_H * (BEV_W / 16));
    int rem = pt % (BEV_H * (BEV_W / 16));
    int y = rem / (BEV_W / 16);
    int x0 = (rem % (BEV_W / 16)) * 16;
    int lx = lane & 15, s = lane >> 4;
    int x = x0 + lx;

    v8f acc[6];
    #pragma unroll
    for (int o = 0; o < 6; ++o) acc[o] = v8f{};

    #pragma unroll
    for (int ks = 0; ks < 27; ++ks) {
        int tap = ks / 3, cib = (ks % 3) * 32;
        int ky = tap / 3, kx = tap % 3;
        int yy = y + ky - 1, xx = x + kx - 1;
        v16h bfrag = {};
        if (yy >= 0 && yy < BEV_H && xx >= 0 && xx < BEV_W)
            bfrag = *(const v16h*)(in + (((size_t)(b * BEV_H + yy) * BEV_W + xx) * CCH
                                         + cib + s * 16));
        #pragma unroll
        for (int o = 0; o < 6; ++o) {
            v16h afrag = *(const v16h*)(wp + ((size_t)(o * 27 + ks) * 32 + lane) * 16);
            acc[o] = __builtin_amdgcn_wmma_f32_16x16x32_f16(
                false, afrag, false, bfrag, (short)0, acc[o], false, false);
        }
    }

    #pragma unroll
    for (int o = 0; o < 6; ++o) {
        int ocb = o * 16 + s * 8;
        #pragma unroll
        for (int r = 0; r < 8; ++r) {
            int oc = ocb + r;
            out[(((size_t)b * CCH + oc) * BEV_H + y) * BEV_W + x] =
                fmaxf(acc[o][r] + bias[oc], 0.f);
        }
    }
}

// ---------------------------------------------------------------------------
extern "C" void kernel_launch(void* const* d_in, const int* in_sizes, int n_in,
                              void* d_out, int out_size, void* d_ws, size_t ws_size,
                              hipStream_t stream) {
    const float* feat   = (const float*)d_in[0];
    const float* l2i    = (const float*)d_in[1];
    const float* he     = (const float*)d_in[2];
    const float* de_w1  = (const float*)d_in[3];
    const float* de_b1  = (const float*)d_in[4];
    const float* de_w2  = (const float*)d_in[5];
    const float* de_b2  = (const float*)d_in[6];
    const float* sh_w1  = (const float*)d_in[7];
    const float* sh_b1  = (const float*)d_in[8];
    const float* sh_w2  = (const float*)d_in[9];
    const float* sh_b2  = (const float*)d_in[10];
    const float* c1w    = (const float*)d_in[11];
    const float* bn1g   = (const float*)d_in[12];
    const float* bn1b   = (const float*)d_in[13];
    const float* bn1m   = (const float*)d_in[14];
    const float* bn1v   = (const float*)d_in[15];
    const float* c2w    = (const float*)d_in[16];
    const float* bn2g   = (const float*)d_in[17];
    const float* bn2b   = (const float*)d_in[18];
    const float* bn2m   = (const float*)d_in[19];
    const float* bn2v   = (const float*)d_in[20];

    const size_t NTOK = (size_t)NB * NCELL * NH;     // 614400
    char* ws = (char*)d_ws;
    size_t off = 0;
    auto carve = [&](size_t bytes) -> void* {
        void* p = ws + off;
        off += (bytes + 255) & ~(size_t)255;
        return p;
    };
    _Float16* tokens  = (_Float16*)carve(NTOK * CCH * 2);
    _Float16* sampled = (_Float16*)carve(NTOK * CCH * 2);
    float*    logits  = (float*)carve(NTOK * 4);
    float*    validf  = (float*)carve(NTOK * 4);
    _Float16* bev     = (_Float16*)carve((size_t)NB * NCELL * CCH * 2);
    _Float16* mid     = (_Float16*)carve((size_t)NB * NCELL * CCH * 2);
    _Float16* w1p     = (_Float16*)carve((size_t)6 * 27 * 512 * 2);
    _Float16* w2p     = (_Float16*)carve((size_t)6 * 27 * 512 * 2);
    _Float16* sw1p    = (_Float16*)carve((size_t)3 * 4 * 512 * 2);
    float*    bias1   = (float*)carve(CCH * 4);
    float*    bias2   = (float*)carve(CCH * 4);

    // K0: weight prep/pack
    prep_kernel<<<(6 * 27 * 512 + 255) / 256, 256, 0, stream>>>(
        c1w, bn1g, bn1b, bn1m, bn1v, c2w, bn2g, bn2b, bn2m, bn2v,
        sh_w1, w1p, bias1, w2p, bias2, sw1p);

    // K1: lift + sample (one wave per (b,cell,h))
    lift_sample_kernel<<<(int)(NTOK / 4), 128, 0, stream>>>(
        feat, l2i, he, de_w1, de_b1, de_w2, de_b2, tokens, sampled, validf);

    // K2: score MLP via WMMA (38400 tiles, 4 waves/block)
    score_kernel<<<38400 / 4, 128, 0, stream>>>(
        tokens, sw1p, sh_b1, sh_w2, sh_b2, logits);

    // K3: softmax + fuse -> BEV NHWC f16
    fuse_kernel<<<(NB * NCELL * CCH) / 256, 256, 0, stream>>>(
        logits, validf, sampled, bev);

    // K4/K5: 3x3 convs via WMMA implicit GEMM (one wave per 16-pixel tile,
    // all 96 output channels accumulated in registers: 6 WMMAs per B-frag load)
    const int pixTiles = NB * BEV_H * (BEV_W / 16);   // 12800
    conv1_kernel<<<pixTiles / 4, 128, 0, stream>>>(bev, w1p, bias1, mid);
    conv2_kernel<<<pixTiles / 4, 128, 0, stream>>>(mid, w2p, bias2, (float*)d_out);
}
